// TopKSparseAttention_4389456577374
// MI455X (gfx1250) — matrix operand order repaired and checked by reading, not compiled, not run on hardware
//
#include <hip/hip_runtime.h>
#include <hip/hip_bf16.h>

typedef __attribute__((ext_vector_type(16))) __bf16 v16bf;
typedef __attribute__((ext_vector_type(8)))  float  v8f;
typedef __attribute__((ext_vector_type(4)))  unsigned int v4u;
typedef __attribute__((ext_vector_type(8)))  int v8i;
typedef __attribute__((ext_vector_type(4)))  int v4i;

#define HID   1024
#define NH    16
#define HD    64
#define TOPK  32
#define INV_SCALE 0.125f   // 1/sqrt(64)

#define CHUNK  512
#define NCHUNK 8           // KV = 4096

__device__ __forceinline__ unsigned umax_u(unsigned a, unsigned b) { return a > b ? a : b; }

// Monotonic float->uint key, top 20 bits of ordered value + 12-bit kv index.
__device__ __forceinline__ unsigned monokey(float f, int idx) {
    unsigned u = __float_as_uint(f);
    u = (u & 0x80000000u) ? ~u : (u | 0x80000000u);
    return (u & 0xFFFFF000u) | (unsigned)idx;
}

// A-fragment (16x32 bf16): lane half h covers K run [16h, 16h+16).
__device__ __forceinline__ v16bf load_afrag(const float* __restrict__ Arow, int kk, int half) {
    const float4* p0 = (const float4*)(Arow + kk + 16 * half);
    const float4* p1 = (const float4*)(Arow + kk + 16 * half + 8);
    float4 x0 = p0[0], x1 = p0[1], y0 = p1[0], y1 = p1[1];
    float t[16] = { x0.x, x0.y, x0.z, x0.w, x1.x, x1.y, x1.z, x1.w,
                    y0.x, y0.y, y0.z, y0.w, y1.x, y1.y, y1.z, y1.w };
    v16bf r;
#pragma unroll
    for (int e = 0; e < 16; ++e) r[e] = (__bf16)t[e];
    return r;
}

// B-fragment from 16 contiguous floats (K run [16h, 16h+16) relative to ptr).
__device__ __forceinline__ v16bf load_bfrag16(const float* p_) {
    const float4* p = (const float4*)p_;
    float4 x0 = p[0], x1 = p[1], x2 = p[2], x3 = p[3];
    float t[16] = { x0.x, x0.y, x0.z, x0.w, x1.x, x1.y, x1.z, x1.w,
                    x2.x, x2.y, x2.z, x2.w, x3.x, x3.y, x3.z, x3.w };
    v16bf r;
#pragma unroll
    for (int e = 0; e < 16; ++e) r[e] = (__bf16)t[e];
    return r;
}

// ---------------------------------------------------------------------------
// C[M,N] = A[M,K] * W[N,K]^T + bias[N]    (fp32 memory, bf16 WMMA math)
// Block = 256 threads (8 waves as 4M x 2N); each wave owns a 32x32 tile
// (2x2 register-blocked WMMA). Block covers 128M x 64N.
// W panel (64 rows x 32 K = 8 KB) staged into LDS by the Tensor Data Mover,
// double-buffered so the TDM fetch of panel kk+32 overlaps WMMA on panel kk.
// grid = (N/64, M/128)
// ---------------------------------------------------------------------------
__global__ void gemm_bias_wmma(const float* __restrict__ A,
                               const float* __restrict__ W,
                               const float* __restrict__ bias,
                               float* __restrict__ C,
                               int M, int N, int K) {
    __shared__ float wpanel[2][64 * 32];   // 2 x 8 KB double buffer

    const int lane = threadIdx.x & 31;
    const int wave = threadIdx.x >> 5;
    const int wm = wave & 3;
    const int wn = wave >> 2;
    const int half = lane >> 4;
    const int l15  = lane & 15;

    int mbase[2];
    const float* Arow[2];
#pragma unroll
    for (int i = 0; i < 2; ++i) {
        mbase[i] = blockIdx.y * 128 + wm * 32 + i * 16;
        Arow[i]  = A + (size_t)(mbase[i] + l15) * K;
    }
    const int nblk = blockIdx.x * 64;

    // ---- TDM descriptor pieces (block-uniform) ----
    const unsigned lds_off[2] = { (unsigned)(unsigned long long)&wpanel[0][0],
                                  (unsigned)(unsigned long long)&wpanel[1][0] };
    const unsigned long long wbase = (unsigned long long)(W + (size_t)nblk * K);
    // D# group1: mask=0, data_size=4B (2), tensor_dim0=K, tensor_dim1=N(rows),
    // tile_dim0=32, tile_dim1=64, dim0_stride=K. (K=N=1024 here.)
    const v8i g1 = { 0x00020000,
                     (int)((unsigned)K << 16),          // tensor_dim0[15:0] @ bits 63:48
                     (int)(((unsigned)K >> 16) | ((unsigned)N << 16)), // dim0 hi | dim1 lo
                     (int)(((unsigned)N >> 16) | (32u << 16)),         // dim1 hi | tile_dim0
                     64,                                 // tile_dim1 (tile_dim2 = 0)
                     K,                                  // dim0_stride lo32
                     0, 0 };
    const v4i gz4 = { 0, 0, 0, 0 };
    const v8i gz8 = { 0, 0, 0, 0, 0, 0, 0, 0 };

    // Issue a TDM load of the 64x32 W panel at column kk into buffer `buf`.
    auto issue_panel = [&](int buf, int kk) {
        unsigned long long ga = wbase + (unsigned long long)kk * 4ull;
        v4u g0 = { 1u,                      // count=1, user mode
                   lds_off[buf],
                   (unsigned)(ga & 0xFFFFFFFFull),
                   (unsigned)((ga >> 32) & 0x01FFFFFFull) | 0x80000000u }; // type=2
        __builtin_amdgcn_tensor_load_to_lds(g0, g1, gz4, gz4, gz8, 0);
    };

    if (wave == 0) {
        issue_panel(0, 0);
        __builtin_amdgcn_s_wait_tensorcnt(0);
    }
    __syncthreads();

    v8f acc[2][2] = {};
    int cur = 0;
    for (int kk = 0; kk < K; kk += 32) {
        const bool more = (kk + 32) < K;
        if (wave == 0 && more) issue_panel(cur ^ 1, kk + 32);
        if (more) {
            __builtin_prefetch(Arow[0] + kk + 128, 0, 0);
            __builtin_prefetch(Arow[1] + kk + 128, 0, 0);
        }

        v16bf af[2], bf_[2];
#pragma unroll
        for (int i = 0; i < 2; ++i) af[i] = load_afrag(Arow[i], kk, half);
#pragma unroll
        for (int j = 0; j < 2; ++j) {
            const int nloc = wn * 32 + j * 16 + l15;          // 0..63
            bf_[j] = load_bfrag16(&wpanel[cur][nloc * 32 + 16 * half]);
        }
#pragma unroll
        for (int i = 0; i < 2; ++i)
#pragma unroll
            for (int j = 0; j < 2; ++j)
                acc[i][j] = __builtin_amdgcn_wmma_f32_16x16x32_bf16(
                    false, af[i], false, bf_[j], (short)0, acc[i][j], false, false);

        if (more) {
            if (wave == 0) __builtin_amdgcn_s_wait_tensorcnt(0);
            __syncthreads();
            cur ^= 1;
        }
    }

#pragma unroll
    for (int j = 0; j < 2; ++j) {
        const int   n  = nblk + wn * 32 + j * 16 + l15;
        const float bv = bias[n];
#pragma unroll
        for (int i = 0; i < 2; ++i)
#pragma unroll
            for (int r = 0; r < 8; ++r) {
                int m = mbase[i] + r + 8 * half;   // C layout: VGPR r -> M = r + 8*half
                C[(size_t)m * N + n] = acc[i][j][r] + bv;
            }
    }
}

// ---------------------------------------------------------------------------
// Fused scores (WMMA) + streaming top-32 + softmax + sparse V gather.
// grid = (Q/16, NH, B), block = 256 (8 waves).
// ---------------------------------------------------------------------------
__global__ void topk_attn(const float* __restrict__ Qp,
                          const float* __restrict__ Kp,
                          const float* __restrict__ Vp,
                          float* __restrict__ Out,
                          int Bn, int Q, int KV) {
    __shared__ float    sc[16][CHUNK];             // 32 KB score tile
    __shared__ unsigned cand[16][NCHUNK * TOPK];   // 16 KB packed candidates

    const int lane = threadIdx.x & 31;
    const int wave = threadIdx.x >> 5;
    const int half = lane >> 4;
    const int l15  = lane & 15;
    const int q0 = blockIdx.x * 16;
    const int h  = blockIdx.y;
    const int b  = blockIdx.z;

    const float* Qbase = Qp + ((size_t)b * Q + q0) * HID + h * HD;
    const float* Kbase = Kp + (size_t)b * KV * HID + h * HD;
    const float* Vbase = Vp + (size_t)b * KV * HID + h * HD;

    // Q-tile A-fragments (M=16, Kdim=64 -> two K=32 frags), constant per block.
    v16bf aq[2];
#pragma unroll
    for (int t = 0; t < 2; ++t)
        aq[t] = load_afrag(Qbase + (size_t)l15 * HID, t * 32, half);

    for (int c = 0; c < NCHUNK; ++c) {
        const int kvbase = c * CHUNK;
        // ---- scores for this chunk: 32 kv-tiles, 4 per wave ----
#pragma unroll
        for (int tt = 0; tt < CHUNK / 16 / 8; ++tt) {
            const int tkv = kvbase + (wave * (CHUNK / 16 / 8) + tt) * 16;
            const float* Krow = Kbase + (size_t)(tkv + l15) * HID;
            v8f acc = {};
#pragma unroll
            for (int t = 0; t < 2; ++t) {
                v16bf bk = load_bfrag16(Krow + t * 32 + 16 * half);
                acc = __builtin_amdgcn_wmma_f32_16x16x32_bf16(
                    false, aq[t], false, bk, (short)0, acc, false, false);
            }
            const int n = (tkv - kvbase) + l15;
#pragma unroll
            for (int r = 0; r < 8; ++r) sc[r + 8 * half][n] = acc[r];
        }
        __syncthreads();

        // ---- per-row top-32 within chunk (wave handles rows wave, wave+8) ----
        for (int rr = 0; rr < 2; ++rr) {
            const int row = wave + rr * 8;
            const int sbase = lane * (CHUNK / 32);
            unsigned lkey = 0;
            for (int i = 0; i < CHUNK / 32; ++i)
                lkey = umax_u(lkey, monokey(sc[row][sbase + i], kvbase + sbase + i));
            unsigned sel = 0;
            for (int r = 0; r < TOPK; ++r) {
                unsigned wmax = lkey;
#pragma unroll
                for (int d = 16; d > 0; d >>= 1)
                    wmax = umax_u(wmax, __shfl_xor(wmax, d, 32));
                if (lane == r) sel = wmax;
                if (lkey == wmax) {   // owner removes winner, rescans its strip
                    int col = (int)(wmax & 0xFFFu) - kvbase;
                    sc[row][col] = -3.0e38f;
                    lkey = 0;
                    for (int i = 0; i < CHUNK / 32; ++i)
                        lkey = umax_u(lkey, monokey(sc[row][sbase + i], kvbase + sbase + i));
                }
            }
            cand[row][c * TOPK + lane] = sel;
        }
        __syncthreads();
    }

    // ---- final top-32 over 256 candidates, softmax, V gather ----
    for (int rr = 0; rr < 2; ++rr) {
        const int row = wave + rr * 8;
        const int sbase = lane * (NCHUNK * TOPK / 32);   // 8 per lane
        unsigned lkey = 0;
        for (int i = 0; i < 8; ++i) lkey = umax_u(lkey, cand[row][sbase + i]);
        unsigned sel = 0;
        for (int r = 0; r < TOPK; ++r) {
            unsigned wmax = lkey;
#pragma unroll
            for (int d = 16; d > 0; d >>= 1)
                wmax = umax_u(wmax, __shfl_xor(wmax, d, 32));
            if (lane == r) sel = wmax;
            if (lkey == wmax) {
                for (int i = 0; i < 8; ++i)
                    if (cand[row][sbase + i] == wmax) cand[row][sbase + i] = 0;
                lkey = 0;
                for (int i = 0; i < 8; ++i) lkey = umax_u(lkey, cand[row][sbase + i]);
            }
        }
        const int idx = (int)(sel & 0xFFFu);

        // exact fp32 score for the selected key
        const float* qrow = Qbase + (size_t)row * HID;
        const float* krow = Kbase + (size_t)idx * HID;
        float s = 0.f;
#pragma unroll
        for (int d = 0; d < HD; d += 4) {
            float4 qv = *(const float4*)(qrow + d);
            float4 kv = *(const float4*)(krow + d);
            s += qv.x * kv.x + qv.y * kv.y + qv.z * kv.z + qv.w * kv.w;
        }
        s *= INV_SCALE;

        float mx = s;
#pragma unroll
        for (int d = 16; d > 0; d >>= 1) mx = fmaxf(mx, __shfl_xor(mx, d, 32));
        float e = __expf(s - mx);
        float sum = e;
#pragma unroll
        for (int d = 16; d > 0; d >>= 1) sum += __shfl_xor(sum, d, 32);
        const float w = e / sum;

        float a0 = 0.f, a1 = 0.f;
        for (int j = 0; j < TOPK; ++j) {
            float wj = __shfl(w, j, 32);
            int   ij = __shfl(idx, j, 32);
            const float* vrow = Vbase + (size_t)ij * HID;
            a0 += wj * vrow[lane];
            a1 += wj * vrow[lane + 32];
        }
        float* orow = Out + ((size_t)b * Q + q0 + row) * HID + h * HD;
        orow[lane]      = a0;
        orow[lane + 32] = a1;
    }
}

extern "C" void kernel_launch(void* const* d_in, const int* in_sizes, int n_in,
                              void* d_out, int out_size, void* d_ws, size_t ws_size,
                              hipStream_t stream) {
    const float* query = (const float*)d_in[0];   // [2,2048,1024]
    const float* key   = (const float*)d_in[1];   // [2,4096,1024]
    const float* value = (const float*)d_in[2];   // [2,4096,1024]
    const float* Wq = (const float*)d_in[3];
    const float* bq = (const float*)d_in[4];
    const float* Wk = (const float*)d_in[5];
    const float* bk = (const float*)d_in[6];
    const float* Wv = (const float*)d_in[7];
    const float* bv = (const float*)d_in[8];
    const float* Wo = (const float*)d_in[9];
    const float* bo = (const float*)d_in[10];

    const int B = 2, Q = 2048, KV = 4096;
    const int MQ = B * Q;     // 4096
    const int MK = B * KV;    // 8192

    float* Qp = (float*)d_ws;                         // 16 MB
    float* Kp = Qp + (size_t)MQ * HID;                // 32 MB
    float* Vp = Kp + (size_t)MK * HID;                // 32 MB
    float* At = Vp + (size_t)MK * HID;                // 16 MB

    dim3 blk(256);
    gemm_bias_wmma<<<dim3(HID / 64, MQ / 128), blk, 0, stream>>>(query, Wq, bq, Qp, MQ, HID, HID);
    gemm_bias_wmma<<<dim3(HID / 64, MK / 128), blk, 0, stream>>>(key,   Wk, bk, Kp, MK, HID, HID);
    gemm_bias_wmma<<<dim3(HID / 64, MK / 128), blk, 0, stream>>>(value, Wv, bv, Vp, MK, HID, HID);

    topk_attn<<<dim3(Q / 16, NH, B), blk, 0, stream>>>(Qp, Kp, Vp, At, B, Q, KV);

    gemm_bias_wmma<<<dim3(HID / 64, MQ / 128), blk, 0, stream>>>(At, Wo, bo, (float*)d_out, MQ, HID, HID);
}
